// QuadraticHeterogenousCrossAttention_23038204576029
// MI455X (gfx1250) — matrix operand order repaired and checked by reading, not compiled, not run on hardware
//
#include <hip/hip_runtime.h>
#include <hip/hip_bf16.h>
#include <math.h>

typedef __attribute__((ext_vector_type(16))) _Float16 v16h;
typedef __attribute__((ext_vector_type(8)))  _Float16 v8h;
typedef __attribute__((ext_vector_type(8)))  float    v8f;

#define WMMA_F16(a,b,c) __builtin_amdgcn_wmma_f32_16x16x32_f16(false,(a),false,(b),(short)0,(c),false,false)

// Problem constants
static constexpr int Bb = 2;      // batches after reshape
static constexpr int Hh = 4;      // heads
static constexpr int Dd = 256;    // model dim
static constexpr int DH = 64;     // head dim
static constexpr int SQ = 1024;   // query length per batch
static constexpr int SK0 = 1024;  // keys source 0
static constexpr int SK1 = 4096;  // keys source 1

// ---------------------------------------------------------------------------
// WMMA GEMM, register-blocked: one wave computes a 16x64 output tile
// (4 accumulators; A-fragment reused across 4 B-fragments per K-step).
// out[r,c] = sum_k A[r,k]*W[k,c] + bias[c].  A f32 or f16; out f16 or f32.
// ---------------------------------------------------------------------------
template <bool A_HALF, bool OUT_F32>
__global__ __launch_bounds__(256)
void gemm_wmma(const void* __restrict__ A_, const float* __restrict__ W,
               const float* __restrict__ bias, void* __restrict__ out_,
               int rows, int K, int N) {
  const int lane   = threadIdx.x & 31;
  const int wave   = threadIdx.x >> 5;
  const int lane15 = lane & 15;
  const int hi     = lane >> 4;
  const int tilesN = N >> 6;                 // 64-wide column tiles
  const int tile   = blockIdx.x * 8 + wave;
  const int tm = tile / tilesN;
  const int n0 = (tile % tilesN) * 64;
  const int arow = tm * 16 + lane15;

  v8f acc[4];
#pragma unroll
  for (int t = 0; t < 4; ++t) acc[t] = (v8f){};

  for (int k0 = 0; k0 < K; k0 += 32) {
    // ---- A fragment: 16x32, M = lane15, halves at k0+hi*8 and k0+16+hi*8
    v16h a;
    if (A_HALF) {
      const _Float16* Ar = (const _Float16*)A_ + (size_t)arow * K;
      v8h lo = *(const v8h*)(Ar + k0 + hi * 8);
      v8h hp = *(const v8h*)(Ar + k0 + 16 + hi * 8);
#pragma unroll
      for (int i = 0; i < 8; ++i) { a[i] = lo[i]; a[8 + i] = hp[i]; }
    } else {
      const float* Ar = (const float*)A_ + (size_t)arow * K;
      const float4* p0 = (const float4*)(Ar + k0 + hi * 8);
      const float4* p1 = (const float4*)(Ar + k0 + 16 + hi * 8);
      float4 x0 = p0[0], x1 = p0[1], y0 = p1[0], y1 = p1[1];
      float f0[8] = {x0.x, x0.y, x0.z, x0.w, x1.x, x1.y, x1.z, x1.w};
      float f1[8] = {y0.x, y0.y, y0.z, y0.w, y1.x, y1.y, y1.z, y1.w};
#pragma unroll
      for (int i = 0; i < 8; ++i) { a[i] = (_Float16)f0[i]; a[8 + i] = (_Float16)f1[i]; }
    }
    // ---- B fragments: 32x16 each; N = lane15, 16 K-rows in table order.
    // Reuse the A fragment across 4 column tiles.
    const float* wr = W + (size_t)k0 * N + n0 + lane15;
#pragma unroll
    for (int t = 0; t < 4; ++t) {
      v16h b;
#pragma unroll
      for (int i = 0; i < 16; ++i) b[i] = (_Float16)wr[(size_t)(i < 8 ? hi * 8 + i : 8 + hi * 8 + i) * N + t * 16];
      acc[t] = WMMA_F16(a, b, acc[t]);
    }
  }
  // ---- store: row = tm*16 + i + 8*hi, col = n0 + t*16 + lane15
#pragma unroll
  for (int i = 0; i < 8; ++i) {
    const int r = tm * 16 + i + 8 * hi;
#pragma unroll
    for (int t = 0; t < 4; ++t) {
      const int c = n0 + t * 16 + lane15;
      const float v = acc[t][i] + bias[c];
      if (OUT_F32) ((float*)out_)[(size_t)r * N + c] = v;
      else         ((_Float16*)out_)[(size_t)r * N + c] = (_Float16)v;
    }
  }
}

// ---------------------------------------------------------------------------
// RoPE + head rearrange.  in: f16 [Bb][S][256]
// ktLayout==0 -> out [Bb][Hh][S][64]   (Q / V)
// ktLayout==1 -> out [Bb][Hh][64][S]   (K transposed for score B-fragments)
// ---------------------------------------------------------------------------
__global__ void rope_rearrange(const _Float16* __restrict__ in,
                               _Float16* __restrict__ out,
                               int S, int nodesPerT, int doRope, int ktLayout) {
  const int idx = blockIdx.x * blockDim.x + threadIdx.x;
  const int total = Bb * S * 128;           // one thread per (even,odd) pair
  if (idx >= total) return;
  const int c2 = idx & 127;
  const int s  = (idx >> 7) % S;
  const int b  = idx / (S * 128);
  const int h  = c2 >> 5;                   // head
  const int p  = c2 & 31;                   // pair index within head
  const int d  = 2 * p;                     // even element within head

  const size_t ib = ((size_t)b * S + s) * Dd + h * DH + d;
  float t1 = (float)in[ib];
  float t2 = (float)in[ib + 1];
  float r0 = t1, r1 = t2;
  if (doRope) {
    const int t = s / nodesPerT;
    const float freq = __powf(10000.0f, -(float)(2 * p) / 64.0f);
    const float ang = (float)t * freq;
    float sn, cs;
    __sincosf(ang, &sn, &cs);
    r0 = t1 * cs - t2 * sn;
    r1 = t1 * sn + t2 * cs;
  }
  if (ktLayout) {
    const size_t ob = (((size_t)(b * Hh + h) * DH + d) * S) + s;
    out[ob]     = (_Float16)r0;
    out[ob + S] = (_Float16)r1;
  } else {
    const size_t ob = (((size_t)(b * Hh + h) * S + s) * DH) + d;
    out[ob]     = (_Float16)r0;
    out[ob + 1] = (_Float16)r1;
  }
}

// ---------------------------------------------------------------------------
// Flash attention: one wave per 16-row q-tile, streaming 32 keys/iteration.
// Q:[Bb][Hh][SQ][64] f16, Kt:[Bb][Hh][64][SK] f16, V:[Bb][Hh][SK][64] f16.
// Writes combined f16 [Bb][SQ][512] at column offset coloff + h*64.
// ---------------------------------------------------------------------------
__global__ __launch_bounds__(256)
void attention_wmma(const _Float16* __restrict__ Q,
                    const _Float16* __restrict__ Kt,
                    const _Float16* __restrict__ V,
                    _Float16* __restrict__ comb,
                    int SK, int coloff) {
  __shared__ _Float16 lds[8][16 * 32];      // per-wave P staging tile (1KB)
  const int lane   = threadIdx.x & 31;
  const int wave   = threadIdx.x >> 5;
  const int lane15 = lane & 15;
  const int hi     = lane >> 4;
  const int tile = blockIdx.x * 8 + wave;   // 512 tiles total
  const int qt = tile & 63;
  const int h  = (tile >> 6) & 3;
  const int b  = tile >> 8;
  const float scale = 0.125f;               // 1/sqrt(64)

  // ---- Q fragments (fixed per tile): dh chunks [0,32) and [32,64)
  const _Float16* qrow = Q + (((size_t)(b * Hh + h) * SQ) + qt * 16 + lane15) * DH;
  v16h aq0, aq1;
  {
    v8h q00 = *(const v8h*)(qrow + hi * 8);
    v8h q01 = *(const v8h*)(qrow + 16 + hi * 8);
    v8h q10 = *(const v8h*)(qrow + 32 + hi * 8);
    v8h q11 = *(const v8h*)(qrow + 48 + hi * 8);
#pragma unroll
    for (int i = 0; i < 8; ++i) {
      aq0[i] = q00[i]; aq0[8 + i] = q01[i];
      aq1[i] = q10[i]; aq1[8 + i] = q11[i];
    }
  }

  float mrow[8], lrow[8];
  v8f acc[4];
#pragma unroll
  for (int i = 0; i < 8; ++i) { mrow[i] = -1e30f; lrow[i] = 0.0f; }
#pragma unroll
  for (int t = 0; t < 4; ++t) acc[t] = (v8f){};

  const _Float16* kbase = Kt + (size_t)(b * Hh + h) * DH * SK;
  const _Float16* vbase0 = V + (size_t)(b * Hh + h) * SK * DH;
  _Float16* P = lds[wave];

  for (int kc = 0; kc < SK; kc += 32) {
    // ---- scores: two 16-key tiles, K-dim = DH in two 32-chunks
    v8f sc[2];
#pragma unroll
    for (int n = 0; n < 2; ++n) {
      v16h bk0, bk1;
      const _Float16* kr0 = kbase + kc + n * 16 + lane15;        // key = lane15, dh in table order
      const _Float16* kr1 = kbase + (size_t)32 * SK + kc + n * 16 + lane15; // key = lane15, dh = 32+table order
#pragma unroll
      for (int i = 0; i < 16; ++i) { const size_t kk = (size_t)(i < 8 ? hi * 8 + i : 8 + hi * 8 + i) * SK; bk0[i] = kr0[kk]; bk1[i] = kr1[kk]; }
      v8f s = {};
      s = WMMA_F16(aq0, bk0, s);
      s = WMMA_F16(aq1, bk1, s);
#pragma unroll
      for (int i = 0; i < 8; ++i) s[i] *= scale;
      sc[n] = s;
    }
    // ---- online softmax per row (row = i + 8*hi; 16 lanes hold its cols)
#pragma unroll
    for (int i = 0; i < 8; ++i) {
      float cm = fmaxf(sc[0][i], sc[1][i]);
      cm = fmaxf(cm, __shfl_xor(cm, 1));
      cm = fmaxf(cm, __shfl_xor(cm, 2));
      cm = fmaxf(cm, __shfl_xor(cm, 4));
      cm = fmaxf(cm, __shfl_xor(cm, 8));
      const float newm = fmaxf(mrow[i], cm);
      const float alpha = __expf(mrow[i] - newm);
      const float p0 = __expf(sc[0][i] - newm);
      const float p1 = __expf(sc[1][i] - newm);
      float rs = p0 + p1;
      rs += __shfl_xor(rs, 1);
      rs += __shfl_xor(rs, 2);
      rs += __shfl_xor(rs, 4);
      rs += __shfl_xor(rs, 8);
      lrow[i] = lrow[i] * alpha + rs;
      mrow[i] = newm;
#pragma unroll
      for (int t = 0; t < 4; ++t) acc[t][i] *= alpha;
      const int r = i + 8 * hi;
      P[r * 32 + lane15]      = (_Float16)p0;
      P[r * 32 + 16 + lane15] = (_Float16)p1;
    }
    __syncthreads();
    // ---- reload P as A-fragment (M = lane15, 32 key cols)
    v16h ap;
    {
      const _Float16* pr = P + lane15 * 32;
      v8h plo = *(const v8h*)(pr + hi * 8);
      v8h phi = *(const v8h*)(pr + 16 + hi * 8);
#pragma unroll
      for (int i = 0; i < 8; ++i) { ap[i] = plo[i]; ap[8 + i] = phi[i]; }
    }
    // ---- AV: B-fragment col = dh lane15, 16 keys in table order
    const _Float16* vr = vbase0 + (size_t)kc * DH + lane15;
#pragma unroll
    for (int t = 0; t < 4; ++t) {
      v16h bv;
#pragma unroll
      for (int i = 0; i < 16; ++i) bv[i] = vr[(size_t)(i < 8 ? hi * 8 + i : 8 + hi * 8 + i) * DH + t * 16];
      acc[t] = WMMA_F16(ap, bv, acc[t]);
    }
    __syncthreads();
  }

  // ---- epilogue: normalize and write combined [Bb][SQ][512]
#pragma unroll
  for (int i = 0; i < 8; ++i) {
    const float inv = 1.0f / lrow[i];
    const int r = qt * 16 + i + 8 * hi;
    const size_t base = ((size_t)b * SQ + r) * (2 * Dd) + coloff + h * DH;
#pragma unroll
    for (int t = 0; t < 4; ++t)
      comb[base + t * 16 + lane15] = (_Float16)(acc[t][i] * inv);
  }
}

// ---------------------------------------------------------------------------
extern "C" void kernel_launch(void* const* d_in, const int* in_sizes, int n_in,
                              void* d_out, int out_size, void* d_ws, size_t ws_size,
                              hipStream_t stream) {
  const float* x0   = (const float*)d_in[0];
  const float* ea   = (const float*)d_in[1];
  const float* qd   = (const float*)d_in[2];
  const float* Wq   = (const float*)d_in[3];
  const float* bq   = (const float*)d_in[4];
  const float* Wk0  = (const float*)d_in[5];
  const float* bk0  = (const float*)d_in[6];
  const float* Wv0  = (const float*)d_in[7];
  const float* bv0  = (const float*)d_in[8];
  const float* Wk1  = (const float*)d_in[9];
  const float* bk1  = (const float*)d_in[10];
  const float* Wv1  = (const float*)d_in[11];
  const float* bv1  = (const float*)d_in[12];
  const float* Wo   = (const float*)d_in[13];
  const float* bo   = (const float*)d_in[14];
  float* out = (float*)d_out;

  // workspace layout (bytes)
  char* ws = (char*)d_ws;
  const size_t MB = 1024 * 1024;
  _Float16* stage = (_Float16*)(ws);              // 4 MB: max 8192x256 f16
  _Float16* Qr    = (_Float16*)(ws + 4 * MB);     // 1 MB: [2][4][1024][64]
  _Float16* Kt0   = (_Float16*)(ws + 5 * MB);     // 1 MB: [2][4][64][1024]
  _Float16* V0    = (_Float16*)(ws + 6 * MB);     // 1 MB
  _Float16* Kt1   = (_Float16*)(ws + 7 * MB);     // 4 MB: [2][4][64][4096]
  _Float16* V1    = (_Float16*)(ws + 11 * MB);    // 4 MB
  _Float16* comb  = (_Float16*)(ws + 15 * MB);    // 2 MB: [2][1024][512]

  auto gemm_f32 = [&](const float* A, const float* W, const float* bias,
                      _Float16* O, int rows, int K, int N) {
    const int tiles = (rows / 16) * (N / 64);     // 16x64 tiles per wave
    gemm_wmma<false, false><<<tiles / 8, 256, 0, stream>>>(A, W, bias, O, rows, K, N);
  };
  auto rearr = [&](const _Float16* in, _Float16* o, int S, int NT, int rope, int kt) {
    const int total = Bb * S * 128;
    rope_rearrange<<<(total + 255) / 256, 256, 0, stream>>>(in, o, S, NT, rope, kt);
  };

  // 1) Q projection + RoPE
  gemm_f32(qd, Wq, bq, stage, Bb * SQ, Dd, Dd);
  rearr(stage, Qr, SQ, 128, 1, 0);
  // 2) K0 projection + RoPE (transposed layout)
  gemm_f32(x0, Wk0, bk0, stage, Bb * SK0, Dd, Dd);
  rearr(stage, Kt0, SK0, 128, 1, 1);
  // 3) V0 projection
  gemm_f32(x0, Wv0, bv0, stage, Bb * SK0, Dd, Dd);
  rearr(stage, V0, SK0, 128, 0, 0);
  // 4) K1 projection + RoPE (transposed)
  gemm_f32(ea, Wk1, bk1, stage, Bb * SK1, Dd, Dd);
  rearr(stage, Kt1, SK1, 512, 1, 1);
  // 5) V1 projection
  gemm_f32(ea, Wv1, bv1, stage, Bb * SK1, Dd, Dd);
  rearr(stage, V1, SK1, 512, 0, 0);

  // 6) attention over both sources -> combined [2][1024][512]
  attention_wmma<<<64, 256, 0, stream>>>(Qr, Kt0, V0, comb, SK0, 0);
  attention_wmma<<<64, 256, 0, stream>>>(Qr, Kt1, V1, comb, SK1, Dd);

  // 7) output projection: combined(f16) @ Wo + bo -> f32 out
  {
    const int rows = Bb * SQ, K = 2 * Dd, N = Dd;
    const int tiles = (rows / 16) * (N / 64);
    gemm_wmma<true, true><<<tiles / 8, 256, 0, stream>>>(comb, Wo, bo, out, rows, K, N);
  }
}
